// F_x_t_43782896615448
// MI455X (gfx1250) — matrix operand order repaired and checked by reading, not compiled, not run on hardware
//
#include <hip/hip_runtime.h>

// ---------------------------------------------------------------------------
// CDNA5 (gfx1250) WMMA implementation of the conv+attention UNet block.
// NHWC internal layout, bf16 operands, f32 accumulation
// (v_wmma_f32_16x16x32_bf16). Flash-style attention (scores stay in LDS).
// 1x1-GEMM weights staged into LDS by the Tensor Data Mover (TENSORcnt).
// 3x3 conv: fully unrolled 36-step K-schedule, 2-stage register pipeline,
// branch-free border handling (clamped loads + cndmask selects).
// ---------------------------------------------------------------------------

typedef __attribute__((ext_vector_type(16))) __bf16 v16bf;
typedef __attribute__((ext_vector_type(8)))  float  v8f;
typedef __attribute__((ext_vector_type(4)))  unsigned int u32x4;
typedef __attribute__((ext_vector_type(8)))  int i32x8;
typedef __attribute__((ext_vector_type(4)))  int i32x4;

union Frag16 {
  v16bf v;
  uint4 q[2];
};

__device__ __forceinline__ unsigned short f2bf(float f) {
  unsigned int u = __float_as_uint(f);
  u += 0x7FFFu + ((u >> 16) & 1u);           // round-to-nearest-even
  return (unsigned short)(u >> 16);
}

__device__ __forceinline__ v8f wmma_bf16(const Frag16& a, const Frag16& b, v8f c) {
  return __builtin_amdgcn_wmma_f32_16x16x32_bf16(false, a.v, false, b.v,
                                                 (short)0, c, false, false);
}

#define ZV8 ((v8f){0.f,0.f,0.f,0.f,0.f,0.f,0.f,0.f})

// ---------------------------------------------------------------------------
// Weight prep: conv3x3 OIHW f32 -> [O][9][C] bf16 (tap-major K ordering)
// ---------------------------------------------------------------------------
__global__ void k_cvt_w3(const float* __restrict__ w, unsigned short* __restrict__ out) {
  int i = blockIdx.x * 256 + threadIdx.x;          // 128*9*128 = 147456
  if (i >= 128 * 9 * 128) return;
  int c = i & 127;
  int tap = (i >> 7) % 9;
  int o = i / (9 * 128);
  out[i] = f2bf(w[(o * 128 + c) * 9 + tap]);
}

__global__ void k_cvt_w1(const float* __restrict__ w, unsigned short* __restrict__ out) {
  int i = blockIdx.x * 256 + threadIdx.x;          // 128*128 = 16384
  if (i >= 128 * 128) return;
  out[i] = f2bf(w[i]);
}

// t-projection: tb[b][o] = sum_k t[b][k]*W[o][k] + bias[o]   (64x128, K=64)
__global__ void k_tproj(const float* __restrict__ t, const float* __restrict__ W,
                        const float* __restrict__ bias, float* __restrict__ out) {
  int i = blockIdx.x * 256 + threadIdx.x;          // 8192
  if (i >= 64 * 128) return;
  int o = i & 127, b = i >> 7;
  float s = bias[o];
  for (int k = 0; k < 64; ++k) s += t[b * 64 + k] * W[o * 64 + k];
  out[i] = s;
}

// NCHW f32 -> NHWC bf16
__global__ void k_nchw_to_nhwc(const float* __restrict__ x, unsigned short* __restrict__ out) {
  int i = blockIdx.x * 256 + threadIdx.x;          // 8388608
  int c = i & 127;
  int hw = (i >> 7) & 1023;
  int b = i >> 17;
  out[i] = f2bf(x[(b * 128 + c) * 1024 + hw]);
}

// NHWC f32 -> NCHW f32 (final output)
__global__ void k_nhwc_to_nchw(const float* __restrict__ x, float* __restrict__ out) {
  int i = blockIdx.x * 256 + threadIdx.x;          // 8388608, indexed by output
  int hw = i & 1023;
  int c = (i >> 10) & 127;
  int b = i >> 17;
  out[i] = x[(b * 1024 + hw) * 128 + c];
}

// ---------------------------------------------------------------------------
// LayerNorm over C (contiguous in NHWC): one wave per row of 128 floats.
// ---------------------------------------------------------------------------
__global__ __launch_bounds__(256) void k_ln(const float* __restrict__ x,
                                            const float* __restrict__ lw,
                                            const float* __restrict__ lb,
                                            unsigned short* __restrict__ out,
                                            int relu) {
  int wave = threadIdx.x >> 5, lane = threadIdx.x & 31;
  int row = blockIdx.x * 8 + wave;                 // 65536 rows
  const float4 v = *reinterpret_cast<const float4*>(x + row * 128 + lane * 4);
  float s = v.x + v.y + v.z + v.w;
  for (int off = 16; off > 0; off >>= 1) s += __shfl_xor(s, off, 32);
  float mu = s * (1.f / 128.f);
  float d0 = v.x - mu, d1 = v.y - mu, d2 = v.z - mu, d3 = v.w - mu;
  float sq = d0 * d0 + d1 * d1 + d2 * d2 + d3 * d3;
  for (int off = 16; off > 0; off >>= 1) sq += __shfl_xor(sq, off, 32);
  float rstd = rsqrtf(sq * (1.f / 128.f) + 1e-5f);
  int c = lane * 4;
  const float4 wv = *reinterpret_cast<const float4*>(lw + c);
  const float4 bv = *reinterpret_cast<const float4*>(lb + c);
  float o0 = d0 * rstd * wv.x + bv.x;
  float o1 = d1 * rstd * wv.y + bv.y;
  float o2 = d2 * rstd * wv.z + bv.z;
  float o3 = d3 * rstd * wv.w + bv.w;
  if (relu) {
    o0 = fmaxf(o0, 0.f); o1 = fmaxf(o1, 0.f);
    o2 = fmaxf(o2, 0.f); o3 = fmaxf(o3, 0.f);
  }
  unsigned short* p = out + row * 128 + c;
  p[0] = f2bf(o0); p[1] = f2bf(o1); p[2] = f2bf(o2); p[3] = f2bf(o3);
}

// ---------------------------------------------------------------------------
// 3x3 conv as implicit GEMM: M=BHW rows, N=128 out channels, K=9*128.
// Wave = 16 M rows x all 128 N; 8 waves/block -> 512 blocks.
// 36 K-steps (tap x cblk) fully unrolled with a 2-stage register pipeline;
// border taps use clamped addresses + cndmask zero-select (no EXEC branches).
// ---------------------------------------------------------------------------
__device__ __forceinline__ void conv_load_step(
    const unsigned short* const (&aptr)[9], unsigned vbits,
    const unsigned short* __restrict__ Wt,
    int tap, int cb, int half, int r15,
    Frag16& af, Frag16 (&bf)[8]) {
  int c0 = cb * 32;
  const uint4 z = make_uint4(0u, 0u, 0u, 0u);
  bool valid = (vbits >> tap) & 1u;
  uint4 t0 = *reinterpret_cast<const uint4*>(aptr[tap] + c0 + half * 16);
  uint4 t1 = *reinterpret_cast<const uint4*>(aptr[tap] + c0 + half * 16 + 8);
  af.q[0] = valid ? t0 : z;
  af.q[1] = valid ? t1 : z;
  int kbase = tap * 128 + c0 + half * 16;
#pragma unroll
  for (int nt = 0; nt < 8; ++nt) {
    const unsigned short* brow = Wt + (nt * 16 + r15) * 1152 + kbase;
    bf[nt].q[0] = *reinterpret_cast<const uint4*>(brow);
    bf[nt].q[1] = *reinterpret_cast<const uint4*>(brow + 8);
  }
}

__device__ __forceinline__ void mma8(const Frag16& af, const Frag16 (&bf)[8],
                                     v8f (&acc)[8]) {
#pragma unroll
  for (int nt = 0; nt < 8; ++nt) acc[nt] = wmma_bf16(af, bf[nt], acc[nt]);
}

__global__ __launch_bounds__(256) void k_conv3x3(
    const unsigned short* __restrict__ A,    // NHWC bf16 activations
    const unsigned short* __restrict__ Wt,   // [O][9][C] bf16
    const float* __restrict__ bias,          // [128] or null
    const float* __restrict__ tb,            // [B][128] or null
    const float* __restrict__ resid,         // NHWC f32 or null
    float* __restrict__ outf,                // NHWC f32 or null
    unsigned short* __restrict__ outbf,      // NHWC bf16 or null
    int relu) {
  int wave = threadIdx.x >> 5, lane = threadIdx.x & 31;
  int half = lane >> 4, r15 = lane & 15;
  int m0 = (blockIdx.x * 8 + wave) * 16;
  int m = m0 + r15;
  int b = m >> 10, hw = m & 1023, h = hw >> 5, w = hw & 31;

  // Per-tap clamped row pointers (always-legal loads) + validity bitmask.
  const unsigned short* aptr[9];
  unsigned vbits = 0;
#pragma unroll
  for (int tap = 0; tap < 9; ++tap) {
    int dh = tap / 3 - 1, dw = tap % 3 - 1;    // constants after unroll
    int hh = h + dh, ww = w + dw;
    unsigned valid = (hh >= 0) & (hh < 32) & (ww >= 0) & (ww < 32);
    vbits |= valid << tap;
    int hc = min(max(hh, 0), 31), wc = min(max(ww, 0), 31);
    aptr[tap] = A + ((((b * 32 + hc) * 32) + wc) << 7);
  }

  v8f acc[8];
#pragma unroll
  for (int i = 0; i < 8; ++i) acc[i] = ZV8;

  Frag16 afA, afB;
  Frag16 bfA[8], bfB[8];
  conv_load_step(aptr, vbits, Wt, 0, 0, half, r15, afA, bfA);
#pragma unroll
  for (int ks = 0; ks < 36; ks += 2) {
    conv_load_step(aptr, vbits, Wt, (ks + 1) >> 2, (ks + 1) & 3,
                   half, r15, afB, bfB);
    mma8(afA, bfA, acc);
    if (ks + 2 < 36)
      conv_load_step(aptr, vbits, Wt, (ks + 2) >> 2, (ks + 2) & 3,
                     half, r15, afA, bfA);
    mma8(afB, bfB, acc);
  }

#pragma unroll
  for (int nt = 0; nt < 8; ++nt) {
    int o = nt * 16 + r15;
    float bo = bias ? bias[o] : 0.f;
#pragma unroll
    for (int r = 0; r < 8; ++r) {
      int me = m0 + r + half * 8;              // C-layout row
      float v = acc[nt][r] + bo;
      if (tb) v += tb[(me >> 10) * 128 + o];
      if (relu) v = fmaxf(v, 0.f);
      if (resid) v += resid[me * 128 + o];
      if (outf) outf[me * 128 + o] = v;
      if (outbf) outbf[me * 128 + o] = f2bf(v);
    }
  }
}

// ---------------------------------------------------------------------------
// 1x1 conv GEMM: [BHW x 128] @ W[128 x 128]^T, K=128.
// Weights (32KB) staged into LDS once per block via the Tensor Data Mover
// (TENSOR_LOAD_TO_LDS, TENSORcnt) with LDS padding of 4 DWORDs per 64 DWORDs
// -> 272B row pitch: 16B aligned for ds_load_b128 and 4-bank rotation per row
// (conflict-free 16-row fragment reads).  Fallback: cooperative copy.
// transpose_out=1 writes bf16 output as [b][c][n] (pre-transposed V).
// ---------------------------------------------------------------------------
#define W1_PITCH 136   // elements per row in LDS (128 + 8 pad)

__global__ __launch_bounds__(256) void k_gemm1x1(
    const unsigned short* __restrict__ A,
    const unsigned short* __restrict__ Wt,   // [O][128] bf16
    const float* __restrict__ bias,
    const float* __restrict__ resid,
    float* __restrict__ outf,
    unsigned short* __restrict__ outbf,
    int transpose_out) {
  __shared__ unsigned short smw[128 * W1_PITCH];   // 34816 B

#if __has_builtin(__builtin_amdgcn_tensor_load_to_lds) && \
    __has_builtin(__builtin_amdgcn_s_wait_tensorcnt)
  if (threadIdx.x < 32) {
    unsigned long long ga = (unsigned long long)Wt;
    unsigned int lds = (unsigned int)(size_t)(void*)smw;
    u32x4 g0;
    g0[0] = 1u;                                   // count=1, user desc
    g0[1] = lds;                                  // lds_addr (bytes)
    g0[2] = (unsigned int)(ga & 0xFFFFFFFFull);   // global_addr[31:0]
    g0[3] = (unsigned int)((ga >> 32) & 0x1FFFFFFull) | (2u << 30); // type=2
    i32x8 g1;
    // data_size=2B (code 1), pad_enable, pad_interval=64DW (code 5),
    // pad_amount=4DW (code 3)
    g1[0] = (1 << 16) | (1 << 20) | (5 << 22) | (3 << 25);
    g1[1] = (128 << 16);          // tensor_dim0[15:0]=128 (bits 79:48)
    g1[2] = (128 << 16);          // tensor_dim1[15:0]=128 (bits 111:80)
    g1[3] = (128 << 16);          // tile_dim0=128        (bits 127:112)
    g1[4] = 128;                  // tile_dim1=128        (bits 143:128)
    g1[5] = 128;                  // tensor_dim0_stride=128 (bits 207:160)
    g1[6] = 0;
    g1[7] = 0;
    i32x4 gz4 = {0, 0, 0, 0};
    i32x8 gz8 = {0, 0, 0, 0, 0, 0, 0, 0};
    // 6-arg form (clang-23 / therock-10.0 toolchain):
    // (group0, group1, group2, group3, extra group, cpol)
    __builtin_amdgcn_tensor_load_to_lds(g0, g1, gz4, gz4, gz8, 0);
    __builtin_amdgcn_s_wait_tensorcnt(0);
  }
  __syncthreads();
#else
  // Cooperative fallback copy with the same padded layout.
#pragma unroll
  for (int it = 0; it < 8; ++it) {
    int q = threadIdx.x + it * 256;               // 2048 x 16B chunks
    int row = q >> 4, c16 = q & 15;
    *reinterpret_cast<uint4*>(smw + row * W1_PITCH + c16 * 8) =
        *reinterpret_cast<const uint4*>(Wt + row * 128 + c16 * 8);
  }
  __syncthreads();
#endif

  int wave = threadIdx.x >> 5, lane = threadIdx.x & 31;
  int half = lane >> 4, r15 = lane & 15;
  int m0 = (blockIdx.x * 8 + wave) * 16;
  const unsigned short* arow = A + ((m0 + r15) << 7);

  v8f acc[8];
#pragma unroll
  for (int i = 0; i < 8; ++i) acc[i] = ZV8;

#pragma unroll
  for (int cb = 0; cb < 4; ++cb) {
    int c0 = cb * 32;
    Frag16 af;
    af.q[0] = *reinterpret_cast<const uint4*>(arow + c0 + half * 16);
    af.q[1] = *reinterpret_cast<const uint4*>(arow + c0 + half * 16 + 8);
    int kbase = c0 + half * 16;
    Frag16 bf[8];
#pragma unroll
    for (int nt = 0; nt < 8; ++nt) {
      const unsigned short* brow = smw + (nt * 16 + r15) * W1_PITCH + kbase;
      bf[nt].q[0] = *reinterpret_cast<const uint4*>(brow);
      bf[nt].q[1] = *reinterpret_cast<const uint4*>(brow + 8);
    }
#pragma unroll
    for (int nt = 0; nt < 8; ++nt) acc[nt] = wmma_bf16(af, bf[nt], acc[nt]);
  }
#pragma unroll
  for (int nt = 0; nt < 8; ++nt) {
    int o = nt * 16 + r15;
    float bo = bias ? bias[o] : 0.f;
#pragma unroll
    for (int r = 0; r < 8; ++r) {
      int me = m0 + r + half * 8;
      float v = acc[nt][r] + bo;
      if (resid) v += resid[me * 128 + o];
      if (outf) outf[me * 128 + o] = v;
      if (outbf) {
        if (transpose_out)
          outbf[((me >> 10) << 17) + o * 1024 + (me & 1023)] = f2bf(v);
        else
          outbf[me * 128 + o] = f2bf(v);
      }
    }
  }
}

// ---------------------------------------------------------------------------
// Flash attention (per batch 1024x1024, d=128, no 1/sqrt(d) per reference).
// 8 waves/block, each wave owns 16 query rows; keys streamed in tiles of 32.
// Scores staged in LDS to cross from WMMA C-layout to A-layout for PV.
// Q,K: [b*1024+n][c] bf16 row-major; VT: [b][c][n] bf16 (pre-transposed).
// All 8 B-fragments of a burst are loaded before the 8 WMMAs.
// ---------------------------------------------------------------------------
__global__ __launch_bounds__(256) void k_flash(
    const unsigned short* __restrict__ Q,
    const unsigned short* __restrict__ K,
    const unsigned short* __restrict__ VT,
    unsigned short* __restrict__ O) {
  __shared__ float sS[8][16][32];
  __shared__ unsigned short sP[8][16][32];
  __shared__ float sM[8][16], sL[8][16], sA[8][16];

  int wave = threadIdx.x >> 5, lane = threadIdx.x & 31;
  int half = lane >> 4, r15 = lane & 15;
  int mq0 = blockIdx.x * 128 + wave * 16;
  int b = mq0 >> 10;

  // Preload all Q A-fragments (K = 128 -> 4 fragments of 32).
  Frag16 qf[4];
  const unsigned short* qp = Q + ((mq0 + r15) << 7);
#pragma unroll
  for (int cb = 0; cb < 4; ++cb) {
    qf[cb].q[0] = *reinterpret_cast<const uint4*>(qp + cb * 32 + half * 16);
    qf[cb].q[1] = *reinterpret_cast<const uint4*>(qp + cb * 32 + half * 16 + 8);
  }

  v8f accO[8];
#pragma unroll
  for (int i = 0; i < 8; ++i) accO[i] = ZV8;

  if (half == 0) { sM[wave][r15] = -1e30f; sL[wave][r15] = 0.f; }
  __syncthreads();

  const unsigned short* Kb = K + (b << 17);
  const unsigned short* Vb = VT + (b << 17);

  for (int kt = 0; kt < 32; ++kt) {
    int j0 = kt * 32;
    // ---- S = Q @ K^T : load all 8 fragments, then burst 8 WMMAs ----
    Frag16 tf[8];
#pragma unroll
    for (int cb = 0; cb < 4; ++cb) {
      int kbase = cb * 32 + half * 16;
      const unsigned short* kr0 = Kb + ((j0 + r15) << 7) + kbase;
      const unsigned short* kr1 = Kb + ((j0 + 16 + r15) << 7) + kbase;
      tf[cb].q[0] = *reinterpret_cast<const uint4*>(kr0);
      tf[cb].q[1] = *reinterpret_cast<const uint4*>(kr0 + 8);
      tf[4 + cb].q[0] = *reinterpret_cast<const uint4*>(kr1);
      tf[4 + cb].q[1] = *reinterpret_cast<const uint4*>(kr1 + 8);
    }
    v8f s0 = ZV8, s1 = ZV8;
#pragma unroll
    for (int cb = 0; cb < 4; ++cb) {
      s0 = wmma_bf16(qf[cb], tf[cb], s0);
      s1 = wmma_bf16(qf[cb], tf[4 + cb], s1);
    }
#pragma unroll
    for (int r = 0; r < 8; ++r) {
      int row = r + half * 8;
      sS[wave][row][r15] = s0[r];
      sS[wave][row][16 + r15] = s1[r];
    }
    __syncthreads();

    // ---- online softmax: 2 lanes per row, 16 cols each ----
    float vals[16];
    float mx = -1e30f;
#pragma unroll
    for (int i = 0; i < 16; ++i) {
      vals[i] = sS[wave][r15][half * 16 + i];
      mx = fmaxf(mx, vals[i]);
    }
    mx = fmaxf(mx, __shfl_xor(mx, 16, 32));
    float mold = sM[wave][r15];
    float mnew = fmaxf(mold, mx);
    float alpha = __expf(mold - mnew);
    float ls = 0.f;
#pragma unroll
    for (int i = 0; i < 16; ++i) {
      float p = __expf(vals[i] - mnew);
      ls += p;
      sP[wave][r15][half * 16 + i] = f2bf(p);
    }
    ls += __shfl_xor(ls, 16, 32);
    if (half == 0) {
      sM[wave][r15] = mnew;
      sL[wave][r15] = sL[wave][r15] * alpha + ls;
      sA[wave][r15] = alpha;
    }
    __syncthreads();

    // ---- rescale accO, then accO += P @ V (batched loads) ----
    float ar[8];
#pragma unroll
    for (int r = 0; r < 8; ++r) ar[r] = sA[wave][r + half * 8];
#pragma unroll
    for (int nt = 0; nt < 8; ++nt)
#pragma unroll
      for (int r = 0; r < 8; ++r) accO[nt][r] *= ar[r];

    Frag16 pf;
    pf.q[0] = *reinterpret_cast<const uint4*>(&sP[wave][r15][half * 16]);
    pf.q[1] = *reinterpret_cast<const uint4*>(&sP[wave][r15][half * 16 + 8]);
#pragma unroll
    for (int nt = 0; nt < 8; ++nt) {
      const unsigned short* vr = Vb + ((nt * 16 + r15) << 10) + j0 + half * 16;
      tf[nt].q[0] = *reinterpret_cast<const uint4*>(vr);
      tf[nt].q[1] = *reinterpret_cast<const uint4*>(vr + 8);
    }
#pragma unroll
    for (int nt = 0; nt < 8; ++nt) accO[nt] = wmma_bf16(pf, tf[nt], accO[nt]);
  }

  float lr[8];
#pragma unroll
  for (int r = 0; r < 8; ++r) lr[r] = 1.f / sL[wave][r + half * 8];
#pragma unroll
  for (int nt = 0; nt < 8; ++nt) {
    int c = nt * 16 + r15;
#pragma unroll
    for (int r = 0; r < 8; ++r) {
      int me = mq0 + r + half * 8;
      O[me * 128 + c] = f2bf(accO[nt][r] * lr[r]);
    }
  }
}

// ---------------------------------------------------------------------------
// Orchestration
// ---------------------------------------------------------------------------
extern "C" void kernel_launch(void* const* d_in, const int* in_sizes, int n_in,
                              void* d_out, int out_size, void* d_ws, size_t ws_size,
                              hipStream_t stream) {
  (void)in_sizes; (void)n_in; (void)out_size; (void)ws_size;
  const float* x      = (const float*)d_in[0];
  const float* t      = (const float*)d_in[1];
  const float* conv_w = (const float*)d_in[2];
  const float* conv_b = (const float*)d_in[3];
  const float* fc_w   = (const float*)d_in[4];
  const float* fc_b   = (const float*)d_in[5];
  const float *r_ln_w[2], *r_ln_b[2], *c1w[2], *c1b[2], *tw[2], *tbp[2], *c2w[2], *c2b[2];
  int base = 6;
  for (int i = 0; i < 2; ++i) {
    r_ln_w[i] = (const float*)d_in[base + 0];
    r_ln_b[i] = (const float*)d_in[base + 1];
    c1w[i]    = (const float*)d_in[base + 2];
    c1b[i]    = (const float*)d_in[base + 3];
    tw[i]     = (const float*)d_in[base + 4];
    tbp[i]    = (const float*)d_in[base + 5];
    c2w[i]    = (const float*)d_in[base + 6];
    c2b[i]    = (const float*)d_in[base + 7];
    base += 8;
  }
  const float *a_ln_w[2], *a_ln_b[2], *qw[2], *kw[2], *vw[2], *ow[2], *ob[2];
  for (int i = 0; i < 2; ++i) {
    a_ln_w[i] = (const float*)d_in[base + 0];
    a_ln_b[i] = (const float*)d_in[base + 1];
    qw[i]     = (const float*)d_in[base + 2];
    kw[i]     = (const float*)d_in[base + 3];
    vw[i]     = (const float*)d_in[base + 4];
    ow[i]     = (const float*)d_in[base + 5];
    ob[i]     = (const float*)d_in[base + 6];
    base += 7;
  }

  // Workspace carve-up (~119.3 MB)
  char* ws = (char*)d_ws;
  float*          cur = (float*)ws;                                  // 33.5 MB f32 NHWC
  unsigned short* abf = (unsigned short*)(ws + 33554432);            // 16.8 MB bf16
  unsigned short* hbf = (unsigned short*)(ws + 50331648);
  unsigned short* qbf = (unsigned short*)(ws + 67108864);
  unsigned short* kbf = (unsigned short*)(ws + 83886080);
  unsigned short* vtb = (unsigned short*)(ws + 100663296);
  unsigned short* w3[5];
  for (int i = 0; i < 5; ++i) w3[i] = (unsigned short*)(ws + 117440512 + (size_t)i * 294912);
  unsigned short* w1[8];
  for (int i = 0; i < 8; ++i) w1[i] = (unsigned short*)(ws + 118915072 + (size_t)i * 32768);
  float* tbv[3];
  for (int i = 0; i < 3; ++i) tbv[i] = (float*)(ws + 119177216 + (size_t)i * 32768);
  unsigned short* obf = abf;  // reuse: abf is dead once q/k/v GEMMs finish

  // --- prep: weights, time projections, layout convert ---
  k_cvt_w3<<<576, 256, 0, stream>>>(conv_w, w3[0]);
  k_cvt_w3<<<576, 256, 0, stream>>>(c1w[0], w3[1]);
  k_cvt_w3<<<576, 256, 0, stream>>>(c2w[0], w3[2]);
  k_cvt_w3<<<576, 256, 0, stream>>>(c1w[1], w3[3]);
  k_cvt_w3<<<576, 256, 0, stream>>>(c2w[1], w3[4]);
  k_cvt_w1<<<64, 256, 0, stream>>>(qw[0], w1[0]);
  k_cvt_w1<<<64, 256, 0, stream>>>(kw[0], w1[1]);
  k_cvt_w1<<<64, 256, 0, stream>>>(vw[0], w1[2]);
  k_cvt_w1<<<64, 256, 0, stream>>>(ow[0], w1[3]);
  k_cvt_w1<<<64, 256, 0, stream>>>(qw[1], w1[4]);
  k_cvt_w1<<<64, 256, 0, stream>>>(kw[1], w1[5]);
  k_cvt_w1<<<64, 256, 0, stream>>>(vw[1], w1[6]);
  k_cvt_w1<<<64, 256, 0, stream>>>(ow[1], w1[7]);
  k_tproj<<<32, 256, 0, stream>>>(t, fc_w, fc_b, tbv[0]);
  k_tproj<<<32, 256, 0, stream>>>(t, tw[0], tbp[0], tbv[1]);
  k_tproj<<<32, 256, 0, stream>>>(t, tw[1], tbp[1], tbv[2]);
  k_nchw_to_nhwc<<<32768, 256, 0, stream>>>(x, abf);

  // --- initial conv: cur = conv3x3(x) + conv_b + (t@fc^T + fc_b) ---
  k_conv3x3<<<512, 256, 0, stream>>>(abf, w3[0], conv_b, tbv[0],
                                     nullptr, cur, nullptr, 0);

  for (int i = 0; i < 2; ++i) {
    // --- res block ---
    k_ln<<<8192, 256, 0, stream>>>(cur, r_ln_w[i], r_ln_b[i], abf, 1);
    // h = relu(conv1(relu(ln)) + b1 + tnet) -> bf16
    k_conv3x3<<<512, 256, 0, stream>>>(abf, w3[1 + 2 * i], c1b[i], tbv[1 + i],
                                       nullptr, nullptr, hbf, 1);
    // cur = conv2(h) + b2 + cur
    k_conv3x3<<<512, 256, 0, stream>>>(hbf, w3[2 + 2 * i], c2b[i], nullptr,
                                       cur, cur, nullptr, 0);

    // --- attention ---
    k_ln<<<8192, 256, 0, stream>>>(cur, a_ln_w[i], a_ln_b[i], abf, 0);
    k_gemm1x1<<<512, 256, 0, stream>>>(abf, w1[4 * i + 0], nullptr, nullptr,
                                       nullptr, qbf, 0);
    k_gemm1x1<<<512, 256, 0, stream>>>(abf, w1[4 * i + 1], nullptr, nullptr,
                                       nullptr, kbf, 0);
    k_gemm1x1<<<512, 256, 0, stream>>>(abf, w1[4 * i + 2], nullptr, nullptr,
                                       nullptr, vtb, 1);   // V pre-transposed
    k_flash<<<512, 256, 0, stream>>>(qbf, kbf, vtb, obf);
    // cur = o @ Wout^T + out_b + cur
    k_gemm1x1<<<512, 256, 0, stream>>>(obf, w1[4 * i + 3], ob[i], cur,
                                       cur, nullptr, 0);
  }

  k_nhwc_to_nchw<<<32768, 256, 0, stream>>>(cur, (float*)d_out);
}
